// EvoformerBlockV49_46961172414802
// MI455X (gfx1250) — matrix operand order repaired and checked by reading, not compiled, not run on hardware
//
#include <hip/hip_runtime.h>
#include <hip/hip_bf16.h>
#include <math.h>

typedef __attribute__((ext_vector_type(16))) _Float16 v16h;
typedef __attribute__((ext_vector_type(8)))  float    v8f;

// ---------------------------------------------------------------------------
// WMMA helpers (wave32, 16x16x32 f16 -> f32), per CDNA5 ISA 7.12.2
// Fragment-per-lane layouts:
//   A (16m x 32k): lane holds row=lane&15; k = kb..kb+15, kb=(lane>>4)*16
//   B (32k x 16n): lane holds col=lane&15; k = kb..kb+15, kb=(lane>>4)*16
//   C (16m x 16n): lane holds col=lane&15; vgpr r -> row = r + 8*(lane>>4)
// ---------------------------------------------------------------------------
__device__ __forceinline__ v8f wmma_f16(v16h a, v16h b, v8f c) {
  return __builtin_amdgcn_wmma_f32_16x16x32_f16(false, a, false, b, (short)0, c,
                                                false, false);
}

// A fragment from f32 row-major, element (m,k) = p[m*ld + k]
__device__ __forceinline__ v16h load_a_frag(const float* __restrict__ p, int ld) {
  int lane = threadIdx.x & 31;
  int row  = lane & 15;
  int kb   = (lane >> 4) << 4;
  const float* r = p + (size_t)row * ld;
  v16h a;
#pragma unroll
  for (int e = 0; e < 8; ++e) a[e]     = (_Float16)r[kb + e];
#pragma unroll
  for (int e = 0; e < 8; ++e) a[8 + e] = (_Float16)r[kb + 8 + e];
  return a;
}

// B fragment, element (k,n) = p[n*ld + k]   (transposed source, e.g. W[N,K])
__device__ __forceinline__ v16h load_b_frag_t(const float* __restrict__ p, int ld) {
  int lane = threadIdx.x & 31;
  int col  = lane & 15;
  int kb   = (lane >> 4) << 4;
  const float* r = p + (size_t)col * ld + kb;
  v16h b;
#pragma unroll
  for (int e = 0; e < 16; ++e) b[e] = (_Float16)r[e];
  return b;
}

// B fragment, element (k,n) = p[k*ld + n]   (natural row-major source)
__device__ __forceinline__ v16h load_b_frag_n(const float* __restrict__ p, int ld) {
  int lane = threadIdx.x & 31;
  int col  = lane & 15;
  int kb   = (lane >> 4) << 4;
  v16h b;
#pragma unroll
  for (int e = 0; e < 16; ++e) b[e] = (_Float16)p[(size_t)(kb + e) * ld + col];
  return b;
}

// ---------------------------------------------------------------------------
// Fragment pre-packing: f32 -> f16 fragments laid out [tile][lane][16]
// so the GEMM/attention inner loops are pure b128 loads + v_wmma.
// ---------------------------------------------------------------------------
__global__ void pack_a16(const float* __restrict__ A, _Float16* __restrict__ P,
                         int M, int K) {
  int kt = K >> 5;
  size_t total = (size_t)(M >> 4) * kt * 32;
  size_t idx = (size_t)blockIdx.x * blockDim.x + threadIdx.x;
  if (idx >= total) return;
  int lane = (int)(idx & 31);
  size_t t = idx >> 5;
  int ks = (int)(t % kt);
  int mt = (int)(t / kt);
  int row = lane & 15;
  int kb  = (lane >> 4) << 4;
  const float* r = A + (size_t)(mt * 16 + row) * K + ks * 32;
  v16h a;
#pragma unroll
  for (int e = 0; e < 8; ++e) a[e]     = (_Float16)r[kb + e];
#pragma unroll
  for (int e = 0; e < 8; ++e) a[8 + e] = (_Float16)r[kb + 8 + e];
  *((v16h*)P + idx) = a;
}

__global__ void pack_w16(const float* __restrict__ W, _Float16* __restrict__ P,
                         int N, int K) {
  int kt = K >> 5;
  size_t total = (size_t)(N >> 4) * kt * 32;
  size_t idx = (size_t)blockIdx.x * blockDim.x + threadIdx.x;
  if (idx >= total) return;
  int lane = (int)(idx & 31);
  size_t t = idx >> 5;
  int ks = (int)(t % kt);
  int nt = (int)(t / kt);
  int col = lane & 15;
  int kb  = (lane >> 4) << 4;
  const float* r = W + (size_t)(nt * 16 + col) * K + ks * 32 + kb;
  v16h b;
#pragma unroll
  for (int e = 0; e < 16; ++e) b[e] = (_Float16)r[e];
  *((v16h*)P + idx) = b;
}

// ---------------------------------------------------------------------------
// Packed GEMM: C[M,N] = A @ W^T + bias. One wave per 16x64 strip: one A
// fragment feeds 4 WMMA per k-step.
// ---------------------------------------------------------------------------
__global__ void gemm_packed(const _Float16* __restrict__ Ap,
                            const _Float16* __restrict__ Wp,
                            const float* __restrict__ bias, float* __restrict__ C,
                            int M, int N, int K) {
  int kt = K >> 5;
  int mt = blockIdx.y;
  int n0 = blockIdx.x << 6;              // 4 n-tiles
  int lane = threadIdx.x & 31;
  const v16h* ap = (const v16h*)Ap + (size_t)mt * kt * 32 + lane;
  const v16h* wp = (const v16h*)Wp + (size_t)(n0 >> 4) * kt * 32 + lane;
  v8f acc[4] = {};
  for (int ks = 0; ks < kt; ++ks) {
    v16h a = ap[(size_t)ks * 32];
    if (ks + 4 < kt) __builtin_prefetch((const void*)(ap + (size_t)(ks + 4) * 32), 0, 3);
#pragma unroll
    for (int t = 0; t < 4; ++t) {
      v16h b = wp[((size_t)t * kt + ks) * 32];
      acc[t] = wmma_f16(a, b, acc[t]);
    }
  }
  int cc = lane & 15;
  int rb = (lane >> 4) << 3;
#pragma unroll
  for (int t = 0; t < 4; ++t) {
    float bv = bias ? bias[n0 + t * 16 + cc] : 0.0f;
#pragma unroll
    for (int r = 0; r < 8; ++r)
      C[(size_t)(mt * 16 + rb + r) * N + n0 + t * 16 + cc] = acc[t][r] + bv;
  }
}

// ---------------------------------------------------------------------------
// MHA (softmax over keys j). qkv = [batch*L, 3C], out = [batch*L, C].
// One wave per (batch, head, 16-row i block). kpm_keep: nonzero byte = keep.
// ---------------------------------------------------------------------------
__global__ void mha_attn(const float* __restrict__ qkv,
                         const unsigned char* __restrict__ kpm_keep,
                         float* __restrict__ out, int L, int C, int heads,
                         float scale) {
  __shared__ float sc[16 * 256];
  __shared__ float rowm[16], rowz[16];
  int i0 = blockIdx.x << 4;
  int h  = blockIdx.y;
  int bb = blockIdx.z;
  int Ch = C / heads;
  int ld = 3 * C;
  int lane = threadIdx.x & 31;
  int cc = lane & 15, rb = (lane >> 4) << 3;
  const float* base = qkv + (size_t)bb * L * ld;

  v16h aq = load_a_frag(base + (size_t)i0 * ld + h * Ch, ld);
  for (int j0 = 0; j0 < L; j0 += 16) {
    v16h bk = load_b_frag_t(base + (size_t)j0 * ld + C + h * Ch, ld);
    v8f z = {};
    v8f s = wmma_f16(aq, bk, z);
    int j = j0 + cc;
    bool keep = (kpm_keep == nullptr) || (kpm_keep[(size_t)bb * L + j] != 0);
#pragma unroll
    for (int r = 0; r < 8; ++r)
      sc[(rb + r) * 256 + j] = keep ? s[r] * scale : -1e9f;
  }
  __syncthreads();
  if (lane < 16) {
    float m = -3.0e38f;
    for (int j = 0; j < L; ++j) m = fmaxf(m, sc[lane * 256 + j]);
    float zs = 0.f;
    for (int j = 0; j < L; ++j) zs += __expf(sc[lane * 256 + j] - m);
    rowm[lane] = m;
    rowz[lane] = zs;
  }
  __syncthreads();

  v8f o0 = {}, o1 = {};
  int prow = lane & 15;
  int pkb  = (lane >> 4) << 4;
  float pm = rowm[prow], pz = rowz[prow];
  for (int j0 = 0; j0 < L; j0 += 32) {
    v16h p;
#pragma unroll
    for (int e = 0; e < 8; ++e)
      p[e] = (_Float16)(__expf(sc[prow * 256 + j0 + pkb + e] - pm) / pz);
#pragma unroll
    for (int e = 0; e < 8; ++e)
      p[8 + e] = (_Float16)(__expf(sc[prow * 256 + j0 + pkb + 8 + e] - pm) / pz);
    const float* vb = base + (size_t)j0 * ld + 2 * C + h * Ch;
    o0 = wmma_f16(p, load_b_frag_n(vb, ld), o0);
    o1 = wmma_f16(p, load_b_frag_n(vb + 16, ld), o1);
  }
#pragma unroll
  for (int r = 0; r < 8; ++r) {
    size_t orow = (size_t)(bb * L + i0 + rb + r) * C + h * Ch;
    out[orow + cc]      = o0[r];
    out[orow + 16 + cc] = o1[r];
  }
}

// ---------------------------------------------------------------------------
// LayerNorm family (one row per block, blockDim.x == C)
// ---------------------------------------------------------------------------
__device__ __forceinline__ float block_ln(float x, int C, const float* __restrict__ g,
                                          const float* __restrict__ b, float* red) {
  int c = threadIdx.x;
  red[c] = x;
  __syncthreads();
  for (int s = C >> 1; s > 0; s >>= 1) {
    if (c < s) red[c] += red[c + s];
    __syncthreads();
  }
  float mean = red[0] / C;
  __syncthreads();
  float d = x - mean;
  red[c] = d * d;
  __syncthreads();
  for (int s = C >> 1; s > 0; s >>= 1) {
    if (c < s) red[c] += red[c + s];
    __syncthreads();
  }
  float var = red[0] / C;
  __syncthreads();
  return d * rsqrtf(var + 1e-5f) * g[c] + b[c];
}

__global__ void ln_rows(const float* __restrict__ in, float* __restrict__ out,
                        const float* __restrict__ g, const float* __restrict__ b,
                        int C) {
  __shared__ float red[256];
  size_t row = blockIdx.x;
  float x = in[row * C + threadIdx.x];
  out[row * C + threadIdx.x] = block_ln(x, C, g, b, red);
}

__global__ void outer_ln(const float* __restrict__ pair, const float* __restrict__ left,
                         const float* __restrict__ g, const float* __restrict__ b,
                         float* __restrict__ out, int N, int C) {
  __shared__ float red[256];
  size_t row = blockIdx.x;
  int i = (int)(row / N), j = (int)(row % N);
  int c = threadIdx.x;
  float x = pair[row * C + c] + left[(size_t)i * C + c] * left[(size_t)j * C + c];
  out[row * C + c] = block_ln(x, C, g, b, red);
}

__global__ void add_ln(const float* __restrict__ a, const float* __restrict__ bIn,
                       const float* __restrict__ g, const float* __restrict__ b,
                       float* __restrict__ out, int C) {
  __shared__ float red[256];
  size_t row = blockIdx.x;
  int c = threadIdx.x;
  float x = a[row * C + c] + bIn[row * C + c];
  out[row * C + c] = block_ln(x, C, g, b, red);
}

// ---------------------------------------------------------------------------
// Elementwise / reshaping kernels
// ---------------------------------------------------------------------------
__global__ void transpose_sn(const float* __restrict__ in, float* __restrict__ out,
                             int S, int N, int C) {          // [s,n,c] -> [n,s,c]
  size_t idx = (size_t)blockIdx.x * blockDim.x + threadIdx.x;
  size_t total = (size_t)S * N * C;
  if (idx >= total) return;
  int c = (int)(idx % C);
  int n = (int)((idx / C) % N);
  int s = (int)(idx / ((size_t)C * N));
  out[((size_t)n * S + s) * C + c] = in[idx];
}

__global__ void transpose_ns(const float* __restrict__ in, float* __restrict__ out,
                             int S, int N, int C) {          // [n,s,c] -> [s,n,c]
  size_t idx = (size_t)blockIdx.x * blockDim.x + threadIdx.x;
  size_t total = (size_t)S * N * C;
  if (idx >= total) return;
  int c = (int)(idx % C);
  int s = (int)((idx / C) % S);
  int n = (int)(idx / ((size_t)C * S));
  out[((size_t)s * N + n) * C + c] = in[idx];
}

__global__ void mean_s(const float* __restrict__ in, float* __restrict__ out,
                       int S, int N, int C) {                // in [n,s,c] -> out [n,c]
  size_t idx = (size_t)blockIdx.x * blockDim.x + threadIdx.x;
  if (idx >= (size_t)N * C) return;
  int c = (int)(idx % C);
  int n = (int)(idx / C);
  float acc = 0.f;
  for (int s = 0; s < S; ++s) acc += in[((size_t)n * S + s) * C + c];
  out[idx] = acc / (float)S;
}

__global__ void gate_mul(const float* __restrict__ x, const float* __restrict__ gsrc,
                         float* __restrict__ out, size_t total) {
  size_t idx = (size_t)blockIdx.x * blockDim.x + threadIdx.x;
  if (idx >= total) return;
  out[idx] = x[idx] * (1.0f / (1.0f + __expf(-gsrc[idx])));
}

// ---------------------------------------------------------------------------
// Triangle attention fragment packing.
//   qP: [h][it][n][lane][16]  A-layout (rows i, k=c)
//   kP: [h][jt][n][lane][16]  B-T layout (k=c, cols j)
//   vP: [h][ct][jb][n][lane][16] B-N layout (k=j, cols c)
// ---------------------------------------------------------------------------
__global__ void pack_tri_q(const float* __restrict__ q, _Float16* __restrict__ P,
                           int N, int H, int Ch) {
  int NT = N >> 4;
  int Cp = H * Ch;
  size_t total = (size_t)H * NT * N * 32;
  size_t idx = (size_t)blockIdx.x * blockDim.x + threadIdx.x;
  if (idx >= total) return;
  int lane = (int)(idx & 31);
  size_t t = idx >> 5;
  int n  = (int)(t % N); t /= N;
  int it = (int)(t % NT);
  int h  = (int)(t / NT);
  int row = lane & 15;
  int kb  = (lane >> 4) << 4;
  const float* r = q + ((size_t)(it * 16 + row) * N + n) * Cp + h * Ch;
  v16h a;
#pragma unroll
  for (int e = 0; e < 8; ++e) a[e]     = (_Float16)r[kb + e];
#pragma unroll
  for (int e = 0; e < 8; ++e) a[8 + e] = (_Float16)r[kb + 8 + e];
  *((v16h*)P + idx) = a;
}

__global__ void pack_tri_k(const float* __restrict__ k, _Float16* __restrict__ P,
                           int N, int H, int Ch) {
  int NT = N >> 4;
  int Cp = H * Ch;
  size_t total = (size_t)H * NT * N * 32;
  size_t idx = (size_t)blockIdx.x * blockDim.x + threadIdx.x;
  if (idx >= total) return;
  int lane = (int)(idx & 31);
  size_t t = idx >> 5;
  int n  = (int)(t % N); t /= N;
  int jt = (int)(t % NT);
  int h  = (int)(t / NT);
  int col = lane & 15;
  int kb  = (lane >> 4) << 4;
  const float* r = k + ((size_t)(jt * 16 + col) * N + n) * Cp + h * Ch + kb;
  v16h b;
#pragma unroll
  for (int e = 0; e < 16; ++e) b[e] = (_Float16)r[e];
  *((v16h*)P + idx) = b;
}

__global__ void pack_tri_v(const float* __restrict__ v, _Float16* __restrict__ P,
                           int N, int H, int Ch) {
  int NB = N >> 5;
  int Cp = H * Ch;
  size_t total = (size_t)H * 2 * NB * N * 32;
  size_t idx = (size_t)blockIdx.x * blockDim.x + threadIdx.x;
  if (idx >= total) return;
  int lane = (int)(idx & 31);
  size_t t = idx >> 5;
  int n  = (int)(t % N); t /= N;
  int jb = (int)(t % NB); t /= NB;
  int ct = (int)(t & 1);
  int h  = (int)(t >> 1);
  int col = lane & 15;
  int kb  = (lane >> 4) << 4;
  v16h b;
#pragma unroll
  for (int e = 0; e < 16; ++e)
    b[e] = (_Float16)v[((size_t)(jb * 32 + kb + e) * N + n) * Cp + h * Ch + ct * 16 + col];
  *((v16h*)P + idx) = b;
}

// ---------------------------------------------------------------------------
// Triangle pass 1: per (h,i,j) running max / sum-exp over n (softmax axis).
// One wave per (jt, it, h); inner loop: 2 b128 loads per matrix + 1 wmma.
// ---------------------------------------------------------------------------
__global__ void tri_stats(const _Float16* __restrict__ qP, const _Float16* __restrict__ kP,
                          const unsigned char* __restrict__ maskj,
                          float* __restrict__ Mst, float* __restrict__ Zst,
                          int N, int H, float scale) {
  int jt = blockIdx.x, it = blockIdx.y, h = blockIdx.z;
  int NT = N >> 4;
  int lane = threadIdx.x & 31;
  int cc = lane & 15, rb = (lane >> 4) << 3;
  int j = jt * 16 + cc;
  bool keep = (maskj == nullptr) || (maskj[j] != 0);
  const v16h* qp = (const v16h*)qP + ((size_t)(h * NT + it) * N) * 32 + lane;
  const v16h* kp = (const v16h*)kP + ((size_t)(h * NT + jt) * N) * 32 + lane;
  float m[8], z[8];
#pragma unroll
  for (int r = 0; r < 8; ++r) { m[r] = -3.0e38f; z[r] = 0.f; }
  for (int n = 0; n < N; ++n) {
    v8f zz = {};
    v8f s = wmma_f16(qp[(size_t)n * 32], kp[(size_t)n * 32], zz);
#pragma unroll
    for (int r = 0; r < 8; ++r) {
      float sv = keep ? s[r] * scale : -1e9f;
      float mn = fmaxf(m[r], sv);
      z[r] = z[r] * __expf(m[r] - mn) + __expf(sv - mn);
      m[r] = mn;
    }
  }
#pragma unroll
  for (int r = 0; r < 8; ++r) {
    size_t o = ((size_t)h * N + (it * 16 + rb + r)) * N + j;
    Mst[o] = m[r];
    Zst[o] = z[r];
  }
}

// ---------------------------------------------------------------------------
// Triangle pass 2: out[i,n,h,c] = sum_j softmax_n(l)[i,j,n] * v[j,n,h,c].
// Recompute logits, normalize with stats, repack P via LDS, P@V with WMMA.
// ---------------------------------------------------------------------------
__global__ void tri_out_k(const _Float16* __restrict__ qP, const _Float16* __restrict__ kP,
                          const _Float16* __restrict__ vP,
                          const float* __restrict__ Mst, const float* __restrict__ Zst,
                          const unsigned char* __restrict__ maskj,
                          float* __restrict__ out, int N, int H, int Ch,
                          int nPerBlk, float scale) {
  __shared__ float sp[16 * 32];
  int it = blockIdx.x, h = blockIdx.y;
  int n0 = blockIdx.z * nPerBlk;
  int NT = N >> 4;
  int NB = N >> 5;
  int Cp = H * Ch;
  int lane = threadIdx.x & 31;
  int cc = lane & 15, rb = (lane >> 4) << 3;
  int prow = lane & 15, pkb = (lane >> 4) << 4;
  const v16h* qp = (const v16h*)qP + ((size_t)(h * NT + it) * N) * 32 + lane;

  for (int n = n0; n < n0 + nPerBlk; ++n) {
    v16h aq = qp[(size_t)n * 32];
    v8f o0 = {}, o1 = {};
    for (int j0 = 0; j0 < N; j0 += 32) {
      int jt = j0 >> 4;
      const v16h* kp0 = (const v16h*)kP + ((size_t)(h * NT + jt) * N + n) * 32 + lane;
      const v16h* kp1 = (const v16h*)kP + ((size_t)(h * NT + jt + 1) * N + n) * 32 + lane;
      v8f zz = {};
      v8f s0 = wmma_f16(aq, *kp0, zz);
      v8f s1 = wmma_f16(aq, *kp1, zz);
      int jA = j0 + cc, jB = j0 + 16 + cc;
      bool keepA = (maskj == nullptr) || (maskj[jA] != 0);
      bool keepB = (maskj == nullptr) || (maskj[jB] != 0);
#pragma unroll
      for (int r = 0; r < 8; ++r) {
        int i = rb + r;
        size_t oA = ((size_t)h * N + (it * 16 + i)) * N + jA;
        size_t oB = ((size_t)h * N + (it * 16 + i)) * N + jB;
        float svA = keepA ? s0[r] * scale : -1e9f;
        float svB = keepB ? s1[r] * scale : -1e9f;
        sp[i * 32 + cc]      = __expf(svA - Mst[oA]) / Zst[oA];
        sp[i * 32 + 16 + cc] = __expf(svB - Mst[oB]) / Zst[oB];
      }
      __syncthreads();
      v16h p;
#pragma unroll
      for (int e = 0; e < 8; ++e) p[e]     = (_Float16)sp[prow * 32 + pkb + e];
#pragma unroll
      for (int e = 0; e < 8; ++e) p[8 + e] = (_Float16)sp[prow * 32 + pkb + 8 + e];
      int jb = j0 >> 5;
      const v16h* vp0 = (const v16h*)vP + (((size_t)(h * 2 + 0) * NB + jb) * N + n) * 32 + lane;
      const v16h* vp1 = (const v16h*)vP + (((size_t)(h * 2 + 1) * NB + jb) * N + n) * 32 + lane;
      o0 = wmma_f16(p, *vp0, o0);
      o1 = wmma_f16(p, *vp1, o1);
      __syncthreads();
    }
#pragma unroll
    for (int r = 0; r < 8; ++r) {
      size_t orow = ((size_t)(it * 16 + rb + r) * N + n) * Cp + h * Ch;
      out[orow + cc]      = o0[r];
      out[orow + 16 + cc] = o1[r];
    }
  }
}

// ---------------------------------------------------------------------------
// Host-side pipeline
// ---------------------------------------------------------------------------
static inline unsigned ceil_div(size_t a, unsigned b) { return (unsigned)((a + b - 1) / b); }

extern "C" void kernel_launch(void* const* d_in, const int* in_sizes, int n_in,
                              void* d_out, int out_size, void* d_ws, size_t ws_size,
                              hipStream_t stream) {
  (void)in_sizes; (void)n_in; (void)out_size; (void)ws_size;
  const float* msa  = (const float*)d_in[0];
  const float* pair = (const float*)d_in[1];
  const unsigned char* mask = (const unsigned char*)d_in[2];
  const float* row_Wqkv = (const float*)d_in[3];
  const float* row_bqkv = (const float*)d_in[4];
  const float* row_Wo   = (const float*)d_in[5];
  const float* row_bo   = (const float*)d_in[6];
  const float* col_Wqkv = (const float*)d_in[7];
  const float* col_bqkv = (const float*)d_in[8];
  const float* col_Wo   = (const float*)d_in[9];
  const float* col_bo   = (const float*)d_in[10];
  const float* op_W = (const float*)d_in[11];
  const float* op_b = (const float*)d_in[12];
  const float* nm_g = (const float*)d_in[13];
  const float* nm_b = (const float*)d_in[14];
  const float* np_g = (const float*)d_in[15];
  const float* np_b = (const float*)d_in[16];
  const float* tq_W = (const float*)d_in[17];
  const float* tq_b = (const float*)d_in[18];
  const float* tk_W = (const float*)d_in[19];
  const float* tk_b = (const float*)d_in[20];
  const float* tv_W = (const float*)d_in[21];
  const float* tv_b = (const float*)d_in[22];
  const float* tg_W = (const float*)d_in[23];
  const float* tg_b = (const float*)d_in[24];
  const float* to_W = (const float*)d_in[25];
  const float* to_b = (const float*)d_in[26];
  const float* tn_g = (const float*)d_in[27];
  const float* tn_b = (const float*)d_in[28];

  constexpr int S = 32, N = 256, Cs = 256, Cp = 128, Hm = 8, Ht = 4, Ch = 32;
  constexpr int Mtok = S * N;                       // 8192
  const float scale = 0.17677669529663687f;         // 1/sqrt(32)

  float* ws = (float*)d_ws;
  size_t off = 0;
  float* qkv   = ws + off; off += (size_t)Mtok * 3 * Cs;      //  6.29M
  float* attn  = ws + off; off += (size_t)Mtok * Cs;          //  2.10M
  float* bufA  = ws + off; off += (size_t)Mtok * Cs;          //  2.10M
  float* mt    = ws + off; off += (size_t)Mtok * Cs;          //  2.10M
  float* msaT  = ws + off; off += (size_t)Mtok * Cs;          //  2.10M
  float* mavg  = ws + off; off += (size_t)N * Cs;
  float* left  = ws + off; off += (size_t)N * Cp;
  float* ApackF = ws + off; off += (size_t)4194304;           // 8.39M f16
  float* WpackF = ws + off; off += (size_t)98304;             // 196608 f16
  float* pairN = ws + off; off += (size_t)N * N * Cp;
  float* tq    = ws + off; off += (size_t)N * N * Cp;
  float* tk    = ws + off; off += (size_t)N * N * Cp;
  float* tv    = ws + off; off += (size_t)N * N * Cp;
  float* tg    = ws + off; off += (size_t)N * N * Cp;
  float* Mst   = ws + off; off += (size_t)Ht * N * N;
  float* Zst   = ws + off; off += (size_t)Ht * N * N;
  float* triO  = ws + off; off += (size_t)N * N * Cp;
  // f16 pack buffers aliasing stage-A/B scratch (dead by stage D)
  _Float16* Apack = (_Float16*)ApackF;
  _Float16* Wpack = (_Float16*)WpackF;
  _Float16* qPack = (_Float16*)qkv;     // 8.39M f16 fits in qkv (6.29M f32)
  _Float16* kPack = (_Float16*)attn;    // spans attn+bufA (4.19M f32)
  _Float16* vPack = (_Float16*)mt;      // spans mt+msaT (4.19M f32)
  float* gated = tq;                    // tq dead after pass 2
  float* projO = tk;                    // tk dead after pass 2

  float* out_msa  = (float*)d_out;
  float* out_pair = (float*)d_out + (size_t)S * N * Cs;

  dim3 wave(32);
  size_t pairElems = (size_t)N * N * Cp;

  auto packed_gemm = [&](const float* A, const float* W, const float* bias, float* C,
                         int M, int Nn, int K) {
    size_t aTot = (size_t)(M >> 4) * (K >> 5) * 32;
    size_t wTot = (size_t)(Nn >> 4) * (K >> 5) * 32;
    pack_a16<<<ceil_div(aTot, 256), 256, 0, stream>>>(A, Apack, M, K);
    pack_w16<<<ceil_div(wTot, 256), 256, 0, stream>>>(W, Wpack, Nn, K);
    gemm_packed<<<dim3(Nn >> 6, M >> 4), wave, 0, stream>>>(Apack, Wpack, bias, C, M, Nn, K);
  };

  // ---- Stage A: row attention over residue axis ----
  packed_gemm(msa, row_Wqkv, row_bqkv, qkv, Mtok, 3 * Cs, Cs);
  mha_attn<<<dim3(N / 16, Hm, S), wave, 0, stream>>>(qkv, mask, attn, N, Cs, Hm, scale);
  packed_gemm(attn, row_Wo, row_bo, bufA, Mtok, Cs, Cs);
  ln_rows<<<Mtok, Cs, 0, stream>>>(bufA, bufA, nm_g, nm_b, Cs);
  transpose_sn<<<(Mtok * Cs + 255) / 256, 256, 0, stream>>>(bufA, mt, S, N, Cs);

  // ---- Stage B: column attention over MSA axis ----
  packed_gemm(mt, col_Wqkv, col_bqkv, qkv, Mtok, 3 * Cs, Cs);
  mha_attn<<<dim3(S / 16, Hm, N), wave, 0, stream>>>(qkv, nullptr, attn, S, Cs, Hm, scale);
  packed_gemm(attn, col_Wo, col_bo, bufA, Mtok, Cs, Cs);
  ln_rows<<<Mtok, Cs, 0, stream>>>(bufA, msaT, nm_g, nm_b, Cs);
  transpose_ns<<<(Mtok * Cs + 255) / 256, 256, 0, stream>>>(msaT, out_msa, S, N, Cs);

  // ---- Stage C: outer product mean pair update ----
  mean_s<<<(N * Cs + 255) / 256, 256, 0, stream>>>(msaT, mavg, S, N, Cs);
  packed_gemm(mavg, op_W, op_b, left, N, Cp, Cs);
  outer_ln<<<N * N, Cp, 0, stream>>>(pair, left, np_g, np_b, pairN, N, Cp);

  // ---- Stage D: triangle attention ----
  // Projections share one packed-A of pairN.
  {
    size_t aTot = (size_t)((N * N) >> 4) * (Cp >> 5) * 32;
    pack_a16<<<ceil_div(aTot, 256), 256, 0, stream>>>(pairN, Apack, N * N, Cp);
    size_t wTot = (size_t)(Cp >> 4) * (Cp >> 5) * 32;
    const float* Ws[4] = {tq_W, tk_W, tv_W, tg_W};
    const float* bs[4] = {tq_b, tk_b, tv_b, tg_b};
    float* outs[4] = {tq, tk, tv, tg};
    for (int p = 0; p < 4; ++p) {
      pack_w16<<<ceil_div(wTot, 256), 256, 0, stream>>>(Ws[p], Wpack, Cp, Cp);
      gemm_packed<<<dim3(Cp >> 6, (N * N) >> 4), wave, 0, stream>>>(Apack, Wpack, bs[p], outs[p], N * N, Cp, Cp);
    }
  }
  {
    size_t tTot = (size_t)Ht * (N >> 4) * N * 32;
    pack_tri_q<<<ceil_div(tTot, 256), 256, 0, stream>>>(tq, qPack, N, Ht, Ch);
    pack_tri_k<<<ceil_div(tTot, 256), 256, 0, stream>>>(tk, kPack, N, Ht, Ch);
    size_t vTot = (size_t)Ht * 2 * (N >> 5) * N * 32;
    pack_tri_v<<<ceil_div(vTot, 256), 256, 0, stream>>>(tv, vPack, N, Ht, Ch);
  }
  tri_stats<<<dim3(N / 16, N / 16, Ht), wave, 0, stream>>>(qPack, kPack, mask, Mst, Zst, N, Ht, scale);
  tri_out_k<<<dim3(N / 16, Ht, 8), wave, 0, stream>>>(qPack, kPack, vPack, Mst, Zst, mask, triO, N, Ht, Ch, N / 8, scale);
  gate_mul<<<ceil_div(pairElems, 256), 256, 0, stream>>>(triO, tg, gated, pairElems);
  packed_gemm(gated, to_W, to_b, projO, N * N, Cp, Cp);
  add_ln<<<N * N, Cp, 0, stream>>>(pairN, projO, tn_g, tn_b, out_pair, Cp);
}
